// MemoryAugmentedLSTMCell_86174223827709
// MI455X (gfx1250) — matrix operand order repaired and checked by reading, not compiled, not run on hardware
//
#include <hip/hip_runtime.h>

#define B_ROWS   16384
#define IN_DIM   512
#define H_DIM    512
#define M_DIM    4096
#define K1       (IN_DIM + H_DIM)   // 1024
#define N1       (4 * H_DIM)        // 2048
#define TILE_ROWS 16

typedef __attribute__((ext_vector_type(16))) __bf16 v16bf;
typedef __attribute__((ext_vector_type(8)))  __bf16 v8bf;
typedef __attribute__((ext_vector_type(8)))  float  v8f;
typedef __attribute__((ext_vector_type(4)))  float  v4f;

static __device__ __forceinline__ v16bf bf_combine(v8bf lo, v8bf hi) {
  return __builtin_shufflevector(lo, hi, 0,1,2,3,4,5,6,7,8,9,10,11,12,13,14,15);
}

// A fragment (16x32, bf16) from LDS row-major [16][ldk] bf16.
// Contiguous K order: lanes 0-15: M=lane, K=k0+0..15;
//             lanes 16-31: M=lane-16, K=k0+16..31.
static __device__ __forceinline__ v16bf a_frag_lds_bf16(const __bf16* A, int ldk, int k0, int lane) {
  int row = lane & 15;
  int kb  = k0 + ((lane >> 4) << 4);
  const __bf16* p = A + (size_t)row * ldk + kb;
  v8bf lo = *(const v8bf*)(p);        // ds_load_b128
  v8bf hi = *(const v8bf*)(p + 8);    // ds_load_b128
  return bf_combine(lo, hi);
}

// B fragment (32x16, bf16) at a per-lane base pointer (base already includes
// col*ldk and the lane-half K offset). 32 contiguous bytes -> 2 global_load_b128.
static __device__ __forceinline__ v16bf b_frag_at(const __bf16* base, size_t off) {
  v8bf lo = *(const v8bf*)(base + off);
  v8bf hi = *(const v8bf*)(base + off + 8);
  return bf_combine(lo, hi);
}

// ---------------------------------------------------------------------------
// Prep: transpose + convert f32 weights [K][N] -> bf16 WT[N][K] in workspace.
// ---------------------------------------------------------------------------
__global__ void __launch_bounds__(256)
transpose_convert_kernel(const float* __restrict__ src, __bf16* __restrict__ dst,
                         int K, int N)
{
  int total = (K >> 3) * N;
  for (int t = blockIdx.x * blockDim.x + threadIdx.x; t < total;
       t += gridDim.x * blockDim.x) {
    int n  = t / (K >> 3);
    int kb = (t - n * (K >> 3)) << 3;
    v8bf v;
#pragma unroll
    for (int i = 0; i < 8; ++i) v[i] = (__bf16)src[(size_t)(kb + i) * N + n];
    *(v8bf*)(dst + (size_t)n * K + kb) = v;
  }
}

// ---------------------------------------------------------------------------
// Kernel 1: attn = softmax(hidden @ W_attn + b_attn); out_cell = 0.1*(attn@memory)
// Logits/probs tile lives in LDS as bf16 (128KB): ~146KB total -> 2 WGs/WGP.
// ---------------------------------------------------------------------------
__global__ void __launch_bounds__(256)
attn_memory_kernel(const float* __restrict__ hidden,
                   const __bf16* __restrict__ WaT,        // [M][H] bf16
                   const float* __restrict__ b_attn,
                   const __bf16* __restrict__ MemT,       // [H][M] bf16
                   float* __restrict__ out_cell_partial,  // d_out + B*H
                   float* __restrict__ out_attn)          // d_out + 2*B*H
{
  extern __shared__ unsigned char smem_raw[];
  __bf16* Ah   = (__bf16*)smem_raw;                                   // 16*512  bf16 (16KB)
  __bf16* P    = (__bf16*)(smem_raw + (size_t)TILE_ROWS * H_DIM * 2); // 16*4096 bf16 (128KB)
  float*  red  = (float*)(smem_raw + (size_t)TILE_ROWS * H_DIM * 2
                                   + (size_t)TILE_ROWS * M_DIM * 2);  // 256 f32
  float*  red2 = red + 256;

  const int tid  = threadIdx.x;
  const int lane = tid & 31;
  const int wave = tid >> 5;
  const int r0   = blockIdx.x * TILE_ROWS;

  // Stage hidden row-block as bf16 into LDS.
  for (int i = tid; i < TILE_ROWS * H_DIM; i += 256) {
    int r = i >> 9, c = i & (H_DIM - 1);
    Ah[i] = (__bf16)hidden[(size_t)(r0 + r) * H_DIM + c];
  }
  __syncthreads();

  // Cache all 16 K-step A fragments in registers — reused for 32 N tiles.
  v16bf afrag[16];
#pragma unroll
  for (int kk = 0; kk < 16; ++kk) afrag[kk] = a_frag_lds_bf16(Ah, H_DIM, kk * 32, lane);

  // logits = hidden @ W_attn + b_attn  -> bf16 tile in LDS
  for (int t = wave; t < M_DIM / 16; t += 8) {
    int n0  = t * 16;
    int col = n0 + (lane & 15);
    const __bf16* base = WaT + (size_t)col * H_DIM + ((lane >> 4) << 4);
    v8f acc = {};
#pragma unroll
    for (int kk = 0; kk < 16; ++kk) {
      if ((kk & 3) == 0) __builtin_prefetch(base + kk * 32 + 256, 0, 1);
      v16bf bfrag = b_frag_at(base, kk * 32);
      acc = __builtin_amdgcn_wmma_f32_16x16x32_bf16(false, afrag[kk], false, bfrag,
                                                    (short)0, acc, false, false);
    }
    float bias = b_attn[col];
    int rowoff = (lane >> 4) << 3;
#pragma unroll
    for (int r = 0; r < 8; ++r)
      P[(size_t)(rowoff + r) * M_DIM + col] = (__bf16)(acc[r] + bias);
  }
  __syncthreads();

  // Row softmax over bf16 LDS tile; each thread owns a contiguous 256-elem chunk.
  {
    int row = tid >> 4, slot = tid & 15;
    __bf16* L = P + (size_t)row * M_DIM + slot * 256;
    float m = -3.4e38f;
    for (int c = 0; c < 256; c += 8) {
      v8bf v = *(const v8bf*)(L + c);
#pragma unroll
      for (int i = 0; i < 8; ++i) m = fmaxf(m, (float)v[i]);
    }
    red[tid] = m;
    __syncthreads();
    float rm = red[row * 16];
#pragma unroll
    for (int s = 1; s < 16; ++s) rm = fmaxf(rm, red[row * 16 + s]);
    float s = 0.f;
    for (int c = 0; c < 256; c += 8) {
      v8bf v = *(const v8bf*)(L + c);
      v8bf e;
#pragma unroll
      for (int i = 0; i < 8; ++i) { float ef = __expf((float)v[i] - rm); e[i] = (__bf16)ef; s += ef; }
      *(v8bf*)(L + c) = e;
    }
    red2[tid] = s;
    __syncthreads();
    float sum = 0.f;
#pragma unroll
    for (int ss = 0; ss < 16; ++ss) sum += red2[row * 16 + ss];
    float inv = 1.f / sum;
    float* ao = out_attn + (size_t)(r0 + row) * M_DIM + slot * 256;
    for (int c = 0; c < 256; c += 8) {
      v8bf v = *(const v8bf*)(L + c);
      v8bf pr;
      v4f o0, o1;
#pragma unroll
      for (int i = 0; i < 8; ++i) {
        float p = (float)v[i] * inv;
        pr[i] = (__bf16)p;
        if (i < 4) o0[i] = p; else o1[i - 4] = p;
      }
      *(v8bf*)(L + c) = pr;                 // bf16 probs for the next GEMM
      *(v4f*)(ao + c)     = o0;             // f32 attn output (16B stores)
      *(v4f*)(ao + c + 4) = o1;
    }
  }
  __syncthreads();

  // memory_read = attn @ memory_matrix; 4 resident accumulators per wave,
  // each A fragment loaded once per K step.
  {
    v8f acc[4] = {};
    const __bf16* base = MemT + (size_t)(wave * 16 + (lane & 15)) * M_DIM
                              + ((lane >> 4) << 4);
    for (int kk = 0; kk < M_DIM / 32; ++kk) {
      v16bf af = a_frag_lds_bf16(P, M_DIM, kk * 32, lane);
#pragma unroll
      for (int tt = 0; tt < 4; ++tt) {
        v16bf bf = b_frag_at(base + (size_t)tt * 128 * M_DIM, (size_t)kk * 32);
        acc[tt] = __builtin_amdgcn_wmma_f32_16x16x32_bf16(false, af, false, bf,
                                                          (short)0, acc[tt], false, false);
      }
    }
    int rowoff = (lane >> 4) << 3;
#pragma unroll
    for (int tt = 0; tt < 4; ++tt) {
      int col = wave * 16 + tt * 128 + (lane & 15);
#pragma unroll
      for (int r = 0; r < 8; ++r)
        out_cell_partial[(size_t)(r0 + rowoff + r) * H_DIM + col] = 0.1f * acc[tt][r];
    }
  }
}

// ---------------------------------------------------------------------------
// Kernel 2: gates = [input|hidden] @ W_gates + b; fused LSTM elementwise epilogue.
// Accumulator-resident: 16 output tiles per wave (128 VGPRs), each A fragment
// loaded once per K step; B addresses = one base + immediate offsets.
// ---------------------------------------------------------------------------
__global__ void __launch_bounds__(256)
gates_kernel(const float* __restrict__ input_tensor,
             const float* __restrict__ hidden,
             const float* __restrict__ cell,
             const __bf16* __restrict__ WgT,     // [N1][K1] bf16
             const float* __restrict__ b_gates,
             float* __restrict__ out_hidden,     // d_out
             float* __restrict__ out_cell)       // d_out + B*H (holds 0.1*mem_read)
{
  extern __shared__ unsigned char smem_raw[];
  __bf16* Ac    = (__bf16*)smem_raw;                                   // 16*1024 bf16 (32KB)
  float*  gates = (float*)(smem_raw + (size_t)TILE_ROWS * K1 * 2);     // 16*2048 f32 (128KB)

  const int tid  = threadIdx.x;
  const int lane = tid & 31;
  const int wave = tid >> 5;
  const int r0   = blockIdx.x * TILE_ROWS;

  // Stage concat(input, hidden) row-block as bf16.
  for (int i = tid; i < TILE_ROWS * K1; i += 256) {
    int r = i >> 10, c = i & (K1 - 1);
    float v = (c < IN_DIM) ? input_tensor[(size_t)(r0 + r) * IN_DIM + c]
                           : hidden[(size_t)(r0 + r) * H_DIM + (c - IN_DIM)];
    Ac[i] = (__bf16)v;
  }
  __syncthreads();

  // Main GEMM: 16 resident accumulator tiles per wave.
  {
    v8f acc[16] = {};
    const __bf16* base = WgT + (size_t)(wave * 16 + (lane & 15)) * K1
                             + ((lane >> 4) << 4);
    for (int kk = 0; kk < K1 / 32; ++kk) {
      __builtin_prefetch(base + kk * 32 + 128, 0, 1);
      v16bf af = a_frag_lds_bf16(Ac, K1, kk * 32, lane);
#pragma unroll
      for (int t = 0; t < 16; ++t) {
        v16bf bf = b_frag_at(base + (size_t)t * 128 * K1, (size_t)kk * 32);
        acc[t] = __builtin_amdgcn_wmma_f32_16x16x32_bf16(false, af, false, bf,
                                                         (short)0, acc[t], false, false);
      }
    }
    int rowoff = (lane >> 4) << 3;
#pragma unroll
    for (int t = 0; t < 16; ++t) {
      int col = wave * 16 + t * 128 + (lane & 15);
      float bias = b_gates[col];
#pragma unroll
      for (int r = 0; r < 8; ++r)
        gates[(size_t)(rowoff + r) * N1 + col] = acc[t][r] + bias;
    }
  }
  __syncthreads();

  // Fused LSTM elementwise epilogue.
  for (int i = tid; i < TILE_ROWS * H_DIM; i += 256) {
    int r = i >> 9, h = i & (H_DIM - 1);
    const float* g = gates + (size_t)r * N1;
    float ig = 1.f / (1.f + __expf(-g[h]));
    float fg = 1.f / (1.f + __expf(-g[H_DIM + h]));
    float cg = tanhf(g[2 * H_DIM + h]);
    float og = 1.f / (1.f + __expf(-g[3 * H_DIM + h]));
    size_t idx = (size_t)(r0 + r) * H_DIM + h;
    float nc = fg * cell[idx] + ig * cg + out_cell[idx];  // out_cell = 0.1*memory_read
    out_cell[idx]   = nc;
    out_hidden[idx] = og * tanhf(nc);
  }
}

extern "C" void kernel_launch(void* const* d_in, const int* in_sizes, int n_in,
                              void* d_out, int out_size, void* d_ws, size_t ws_size,
                              hipStream_t stream) {
  (void)in_sizes; (void)n_in; (void)out_size; (void)ws_size;
  const float* input_tensor = (const float*)d_in[0];
  const float* hidden       = (const float*)d_in[1];
  const float* cell         = (const float*)d_in[2];
  const float* W_gates      = (const float*)d_in[3];
  const float* b_gates      = (const float*)d_in[4];
  const float* W_attn       = (const float*)d_in[5];
  const float* b_attn       = (const float*)d_in[6];
  const float* memory_mtx   = (const float*)d_in[7];

  float* out        = (float*)d_out;
  float* out_hidden = out;
  float* out_cell   = out + (size_t)B_ROWS * H_DIM;
  float* out_attn   = out + 2ull * (size_t)B_ROWS * H_DIM;

  // Workspace: three transposed bf16 weight matrices (4MB each, L2-resident).
  __bf16* WgT  = (__bf16*)d_ws;                 // [N1][K1]
  __bf16* WaT  = WgT + (size_t)N1 * K1;         // [M][H]
  __bf16* MemT = WaT + (size_t)M_DIM * H_DIM;   // [H][M]

  dim3 block(256);
  transpose_convert_kernel<<<dim3(1024), block, 0, stream>>>(W_gates, WgT, K1, N1);
  transpose_convert_kernel<<<dim3(1024), block, 0, stream>>>(W_attn, WaT, H_DIM, M_DIM);
  transpose_convert_kernel<<<dim3(1024), block, 0, stream>>>(memory_mtx, MemT, M_DIM, H_DIM);

  dim3 grid(B_ROWS / TILE_ROWS);

  size_t smem1 = (size_t)TILE_ROWS * H_DIM * 2      // Ah bf16 (16KB)
               + (size_t)TILE_ROWS * M_DIM * 2      // probs bf16 (128KB)
               + 512 * 4;                           // reduction scratch
  size_t smem2 = (size_t)TILE_ROWS * K1 * 2         // Ac bf16 (32KB)
               + (size_t)TILE_ROWS * N1 * 4;        // gates f32 (128KB)

  attn_memory_kernel<<<grid, block, smem1, stream>>>(hidden, WaT, b_attn, MemT,
                                                     out_cell, out_attn);
  gates_kernel<<<grid, block, smem2, stream>>>(input_tensor, hidden, cell, WgT, b_gates,
                                               out_hidden, out_cell);
}
